// BinaryTreeShConv_28243704938902
// MI455X (gfx1250) — matrix operand order repaired and checked by reading, not compiled, not run on hardware
//
#include <hip/hip_runtime.h>

typedef __attribute__((ext_vector_type(16))) _Float16 v16h;
typedef __attribute__((ext_vector_type(8)))  _Float16 v8h;
typedef __attribute__((ext_vector_type(8)))  float    v8f;
typedef __attribute__((ext_vector_type(4)))  unsigned int u32x4;
typedef __attribute__((ext_vector_type(8)))  int          i32x8;
typedef __attribute__((ext_vector_type(4)))  int          i32x4;

// Problem constants (B=8, V=4096, P=32, C=32, R=2, N=16, OUT=32)
constexpr int kBV    = 32768;   // B*V flat vertices
constexpr int kC     = 32;
constexpr int kP     = 32;
constexpr int kRN    = 32;      // R*N
constexpr int kOUT   = 32;
constexpr int kKTOT  = 1024;    // C*RN
constexpr int kVGRP  = 16;      // vertices per wave
constexpr int kNGRP  = kBV / kVGRP;   // 2048 wave-groups
constexpr int kWAVES = 2;
constexpr int kTHREADS = kWAVES * 32;

// LDS layout (units: _Float16 halves). Row strides are 16B multiples so
// fragment loads can be ds_load_b128.
constexpr int kWStride  = 1032;               // 1024 + 8 pad (2064B rows)
constexpr int kWHalves  = kOUT * kWStride;    // 33024
constexpr int kPtStride = 40;                 // 32 + 8 pad (80B rows)
constexpr int kStageHalves = 32 * kPtStride;  // 1280
constexpr int kYVStride = 1032;               // 32*32 + 8 pad per vertex
constexpr int kYHalves  = kVGRP * kYVStride;  // 16512
constexpr int kWaveHalves = kStageHalves + kYHalves; // 17792

#define HAS_TDM __has_builtin(__builtin_amdgcn_tensor_load_to_lds)

// Assemble a v16h WMMA operand from two 16-byte LDS chunks.
__device__ __forceinline__ v16h frag2(const _Float16* p0, const _Float16* p1) {
  const v8h a = *(const v8h*)p0;
  const v8h b = *(const v8h*)p1;
  v16h r;
#pragma unroll
  for (int i = 0; i < 8; ++i) { r[i] = a[i]; r[i + 8] = b[i]; }
  return r;
}

#if HAS_TDM
// TDM: async-copy one contiguous [P=32 x RN=32] f32 tile (4KB) global -> LDS.
// D# per CDNA5 ISA ch.8: 2D tensor, data_size=4B, tile 1024x1.
// clang-23 builtin arity: (u32x4 g0, i32x8 g1, i32x4 g2, i32x4 g3, i32x8, i32 cpol)
__device__ __forceinline__ void tdm_load_tile(const float* gsrc, unsigned lds_off) {
  const unsigned long long ga = (unsigned long long)(size_t)gsrc;
  u32x4 g0;
  g0[0] = 1u;                                            // count=1, is_restore=0
  g0[1] = lds_off;                                       // lds_addr (bytes)
  g0[2] = (unsigned)ga;                                  // global_addr[31:0]
  g0[3] = (unsigned)((ga >> 32) & 0x1FFFFFFu) | (2u << 30); // addr[56:32], type=2
  i32x8 g1;
  g1[0] = (2 << 16);          // workgroup_mask=0, data_size=2 (4B)
  g1[1] = (int)(1024u << 16); // tensor_dim0 = 1024 (bits 79:48 low half)
  g1[2] = (1 << 16);          // tensor_dim0 hi = 0, tensor_dim1 = 1
  g1[3] = (int)(1024u << 16); // tensor_dim1 hi = 0, tile_dim0 = 1024
  g1[4] = 1;                  // tile_dim1 = 1, tile_dim2 = 0
  g1[5] = 1024;               // tensor_dim0_stride lo
  g1[6] = 0;                  // stride hi / tensor_dim1_stride lo
  g1[7] = 0;
  const i32x4 z4 = {0, 0, 0, 0};                 // 2D tensor: groups 2/3 unused
  const i32x8 z8 = {0, 0, 0, 0, 0, 0, 0, 0};     // unused trailing group
  __builtin_amdgcn_tensor_load_to_lds(g0, g1, z4, z4, z8, 0);
}
#endif

__global__ __launch_bounds__(kTHREADS) void btsh_fused_wmma(
    const float* __restrict__ signal,   // [B,V,C]
    const int*   __restrict__ pidx,     // [B,V,P]
    const float* __restrict__ convk,    // [B,V,P,RN]
    const float* __restrict__ kwt,      // [OUT,C,R,N] == [OUT, KTOT]
    const float* __restrict__ bias,     // [OUT]
    float*       __restrict__ out)      // [B,V,OUT]
{
  __shared__ __align__(16) _Float16 lds[kWHalves + kWAVES * kWaveHalves];
  __shared__ __align__(16) float ldsk[kWAVES][2][kP * kRN]; // TDM conv buffers
  __shared__ float ldsb[kOUT];

  const int tid  = threadIdx.x;
  const int lane = tid & 31;
  const int wave = tid >> 5;
  const int hi   = lane >> 4;   // lane half (selects K sub-range of fragments)
  const int ln   = lane & 15;   // row/col within 16

  // Stage kernel_weights f32 -> f16 into padded LDS rows (once per WG).
  for (int e = tid; e < kOUT * kKTOT; e += kTHREADS) {
    const int row = e >> 10, col = e & (kKTOT - 1);
    lds[row * kWStride + col] = (_Float16)kwt[e];
  }
  if (tid < kOUT) ldsb[tid] = bias[tid];
  __syncthreads();

  _Float16* Wl = lds;
  _Float16* Pt = lds + kWHalves + wave * kWaveHalves;  // patches^T [c][p], f16
  _Float16* Yl = Pt + kStageHalves;                    // Y blocks [v][c][rn], f16

  const int group = blockIdx.x * kWAVES + wave;        // 0..2047

  v8f acc0 = {}; v8f acc1 = {};   // Out rows 0..15 / 16..31 accumulators

#if HAS_TDM
  // Prime the double buffer: DMA conv tile for vertex 0 of this wave's group.
  tdm_load_tile(convk + (size_t)(group * kVGRP) * kKTOT,
                (unsigned)(size_t)&ldsk[wave][0][0]);
#endif

  for (int vv = 0; vv < kVGRP; ++vv) {
    const int u = group * kVGRP + vv;          // flat vertex (b*V + v)
    const int b = u >> 12;                     // V = 4096
    const float* sigb = signal + (size_t)b * 4096 * kC;
    const int myidx = pidx[u * kP + lane];     // idx for p = lane

#if HAS_TDM
    if (vv + 1 < kVGRP) {   // overlap: DMA next tile while computing this one
      tdm_load_tile(convk + (size_t)(u + 1) * kKTOT,
                    (unsigned)(size_t)&ldsk[wave][(vv + 1) & 1][0]);
    }
#else
    {  // fallback staging: plain vector loads into the same LDS buffer
      float* kf = &ldsk[wave][vv & 1][0];
#pragma unroll
      for (int p = 0; p < kP; ++p) kf[p * kRN + lane] =
          convk[(size_t)u * kKTOT + p * kRN + lane];
    }
#endif

    // Gather patches (indexed -> stays on the vector-memory pipe),
    // store transposed as f16: Pt[c][p].
#pragma unroll
    for (int p = 0; p < kP; ++p) {
      const int ip = __shfl(myidx, p, 32);               // broadcast idx[p]
      Pt[lane * kPtStride + p] = (_Float16)sigb[ip * kC + lane];
    }

#if HAS_TDM
    if (vv + 1 < kVGRP) __builtin_amdgcn_s_wait_tensorcnt(1);
    else                __builtin_amdgcn_s_wait_tensorcnt(0);
#endif
    const float* Kf = &ldsk[wave][vv & 1][0];            // conv f32 [p][rn]

    // Step 1: Y[32x32] = patches^T (A) @ conv (B), K = P = 32.
    const int klo = hi << 3;
    const _Float16* ar0 = Pt + (ln +  0) * kPtStride;
    const _Float16* ar1 = Pt + (ln + 16) * kPtStride;
    v16h A0 = frag2(ar0 + klo, ar0 + 16 + klo);
    v16h A1 = frag2(ar1 + klo, ar1 + 16 + klo);

    // B fragments straight from the raw f32 TDM tile (strided rows, cvt f16).
    // lane -> column n; lo lanes k=0..7,16..23, hi lanes k=8..15,24..31.
    v16h B0, B1;
    {
      const float* c0 = Kf + (hi << 3) * kRN + (ln +  0); // column rn = ln
      const float* c1 = Kf + (hi << 3) * kRN + (ln + 16); // column rn = 16+ln
#pragma unroll
      for (int kk = 0; kk < 16; ++kk) {
        B0[kk] = (_Float16)c0[(kk + (kk & 8)) * kRN];
        B1[kk] = (_Float16)c1[(kk + (kk & 8)) * kRN];
      }
    }

    v8f y00 = {}, y01 = {}, y10 = {}, y11 = {};
    y00 = __builtin_amdgcn_wmma_f32_16x16x32_f16(false, A0, false, B0, (short)0, y00, false, false);
    y01 = __builtin_amdgcn_wmma_f32_16x16x32_f16(false, A0, false, B1, (short)0, y01, false, false);
    y10 = __builtin_amdgcn_wmma_f32_16x16x32_f16(false, A1, false, B0, (short)0, y10, false, false);
    y11 = __builtin_amdgcn_wmma_f32_16x16x32_f16(false, A1, false, B1, (short)0, y11, false, false);

    // Spill Y (f32->f16) into this wave's Y block: [vv][c][rn].
    _Float16* yv = Yl + vv * kYVStride;
#pragma unroll
    for (int j = 0; j < 8; ++j) {
      const int c0 = j + hi * 8;
      yv[(c0     ) * kRN + (ln     )] = (_Float16)y00[j];
      yv[(c0     ) * kRN + (ln + 16)] = (_Float16)y01[j];
      yv[(c0 + 16) * kRN + (ln     )] = (_Float16)y10[j];
      yv[(c0 + 16) * kRN + (ln + 16)] = (_Float16)y11[j];
    }
  }

  // Step 2: Out[32x16] = W[32x1024] @ Yblk[1024x16], K chunk kc == channel c.
  const int klo = hi << 3;
#pragma unroll 4
  for (int kc = 0; kc < 32; ++kc) {
    const _Float16* yb = Yl + ln * kYVStride + kc * kRN + (hi << 3);
    v16h Bf = frag2(yb, yb + 16);                      // B: Y[v=ln][kc][rn]
    const _Float16* wr0 = Wl + (ln +  0) * kWStride + kc * 32;
    const _Float16* wr1 = Wl + (ln + 16) * kWStride + kc * 32;
    v16h Af0 = frag2(wr0 + klo, wr0 + 16 + klo);
    v16h Af1 = frag2(wr1 + klo, wr1 + 16 + klo);
    acc0 = __builtin_amdgcn_wmma_f32_16x16x32_f16(false, Af0, false, Bf, (short)0, acc0, false, false);
    acc1 = __builtin_amdgcn_wmma_f32_16x16x32_f16(false, Af1, false, Bf, (short)0, acc1, false, false);
  }

  // Bias + relu + store. D layout: element (m = j + hi*8, n = ln); n = vertex.
  float* op = out + (size_t)(group * kVGRP + ln) * kOUT;
#pragma unroll
  for (int j = 0; j < 8; ++j) {
    const int i0 = j + hi * 8;
    const float r0 = acc0[j] + ldsb[i0];
    const float r1 = acc1[j] + ldsb[16 + i0];
    op[i0]      = r0 > 0.f ? r0 : 0.f;
    op[16 + i0] = r1 > 0.f ? r1 : 0.f;
  }
}

extern "C" void kernel_launch(void* const* d_in, const int* in_sizes, int n_in,
                              void* d_out, int out_size, void* d_ws, size_t ws_size,
                              hipStream_t stream) {
  (void)in_sizes; (void)n_in; (void)out_size; (void)d_ws; (void)ws_size;
  const float* signal = (const float*)d_in[0];
  const int*   pidx   = (const int*)d_in[1];
  const float* convk  = (const float*)d_in[2];
  const float* kwt    = (const float*)d_in[3];
  const float* bias   = (const float*)d_in[4];
  float* out = (float*)d_out;

  dim3 grid(kNGRP / kWAVES);   // 1024 workgroups
  dim3 block(kTHREADS);        // 64 threads = 2 waves
  hipLaunchKernelGGL(btsh_fused_wmma, grid, block, 0, stream,
                     signal, pidx, convk, kwt, bias, out);
}
